// DGANGenerator_3573412791057
// MI455X (gfx1250) — matrix operand order repaired and checked by reading, not compiled, not run on hardware
//
#include <hip/hip_runtime.h>
#include <hip/hip_bf16.h>
#include <math.h>

// ---------------- problem constants ----------------
#define B_   256
#define T_   256
#define H_   256
// D_IN = 80, padded to 96 for K%32==0

// output offsets (floats), reference tuple order
#define O1 0u          // temporal_cont [B,T,8]
#define O2 524288u     // deltas [B,T]
#define O3 589824u     // cont_buf [B,T,8]
#define O4 1114112u    // followup [B]
#define O5 1114368u    // n_v [B]
#define O6 1114624u    // static_cont [B,4]
#define O7 1115648u    // static_cat [B,6]
#define O8 1117184u    // static_cat_soft [B,6]
#define O9 1118720u    // irr [B,T,2]
#define O10 1249792u   // reg [B,T,5]

typedef __bf16 v16bf __attribute__((ext_vector_type(16)));
typedef float  v8f   __attribute__((ext_vector_type(8)));

#define WMMA_BF16(a,b,c) \
  __builtin_amdgcn_wmma_f32_16x16x32_bf16(false,(a),false,(b),(short)0,(c),false,false)

struct __align__(16) US8 { unsigned short h[8]; };
union FragU { US8 u[2]; unsigned short s[16]; v16bf v; };

__device__ __forceinline__ v8f vzero(){
  v8f z;
#pragma unroll
  for (int i=0;i<8;i++) z[i]=0.f;
  return z;
}

__device__ __forceinline__ unsigned short f2bf(float f){
  union { float f; unsigned int u; } v; v.f = f;
  unsigned int u = v.u;
  unsigned int r = (u + 0x7FFFu + ((u >> 16) & 1u)) >> 16;   // RNE
  return (unsigned short)r;
}

// ---- fragment loaders (16-bit A/B, this code's contiguous K order k = 16h + s in both A and B) ----

// A fragment from row-major bf16 global: lane m=row0+(l&15); halves hold contiguous K groups of 16
__device__ __forceinline__ v16bf afrag_g(const unsigned short* A, int lda, int row0, int k0, int lane){
  int m  = row0 + (lane & 15);
  int kb = k0 + ((lane >> 4) << 4);
  const unsigned short* p = A + (size_t)m * lda + kb;
  FragU c; c.u[0] = *(const US8*)p; c.u[1] = *(const US8*)(p + 8);
  return c.v;
}

// A fragment from row-major bf16 LDS tile: two 16B ds loads, no conversion
__device__ __forceinline__ v16bf afrag_lb(const unsigned short* S, int lda, int k0, int lane){
  int m  = lane & 15;
  int kb = k0 + ((lane >> 4) << 4);
  const unsigned short* p = S + m * lda + kb;
  FragU c; c.u[0] = *(const US8*)p; c.u[1] = *(const US8*)(p + 8);
  return c.v;
}

// B fragment from pre-swizzled weights: [ntile][ktile][lane][16 contiguous bf16]
__device__ __forceinline__ v16bf bfrag_w(const unsigned short* Wf, int Kpad, int ntile, int kt, int lane){
  const unsigned short* p = Wf + (((size_t)ntile * (Kpad >> 5) + kt) << 9) + (lane << 4);
  FragU c; c.u[0] = ((const US8*)p)[0]; c.u[1] = ((const US8*)p)[1];
  return c.v;
}

// B fragment where B[k][n] = M[row0+n][col0+k]  (rows of M are columns of B; contiguous K)
__device__ __forceinline__ v16bf bfrag_rows(const unsigned short* M, int ldm, int row0, int col0, int lane){
  const unsigned short* p = M + (size_t)(row0 + (lane & 15)) * ldm + col0 + ((lane >> 4) << 4);
  FragU c; c.u[0] = ((const US8*)p)[0]; c.u[1] = ((const US8*)p)[1];
  return c.v;
}

// B fragment where B[k][n] = M[row0+k][col0+n]  (strided gather, used for V in attention)
__device__ __forceinline__ v16bf bfrag_cols(const unsigned short* M, int ldm, int row0, int col0, int lane){
  FragU c;
  int n = lane & 15, kb = (lane >> 4) << 4;
#pragma unroll
  for (int i=0;i<16;i++) c.s[i] = M[(size_t)(row0 + kb + i) * ldm + col0 + n];
  return c.v;
}

// ---- scalar math / hash RNG (gumbel noise; exact jax bits not reproducible) ----
__device__ __forceinline__ float sigmoidf_(float x){ return 1.f/(1.f+__expf(-x)); }
__device__ __forceinline__ float softplusf_(float x){ return x>20.f ? x : log1pf(__expf(x)); }
__device__ __forceinline__ float urand_(unsigned a, unsigned b, unsigned c){
  unsigned x = a*0x9E3779B1u ^ (b+0x7F4A7C15u)*0x85EBCA77u ^ (c+0x165667B1u)*0xC2B2AE3Du;
  x ^= x>>16; x *= 0x7FEB352Du; x ^= x>>15; x *= 0x846CA68Bu; x ^= x>>16;
  float u = (float)(x>>8) * (1.f/16777216.f);
  return fminf(fmaxf(u, 1e-6f), 1.f-1e-6f);
}
__device__ __forceinline__ float gumbelr(unsigned a, unsigned b, unsigned c){
  return -logf(-logf(urand_(a,b,c)));
}

// ---------------- weight pre-swizzle + f32->bf16 ----------------
__global__ void __launch_bounds__(256) k_wfrag(const float* __restrict__ W,
                                               unsigned short* __restrict__ Wf,
                                               int N, int Korig, int Kpad){
  int idx = blockIdx.x*256 + threadIdx.x;
  if (idx >= N*Kpad) return;
  int tile = idx >> 9, rem = idx & 511;
  int lane = rem >> 4, e = rem & 15;
  int ktiles = Kpad >> 5;
  int nt = tile / ktiles, kt = tile - nt*ktiles;
  int n = (nt<<4) + (lane & 15);
  int k = (kt<<5) + ((lane>>4)<<4) + e;
  float v = (k < Korig) ? W[(size_t)n*Korig + k] : 0.f;
  Wf[idx] = f2bf(v);
}

__global__ void __launch_bounds__(256) k_tobf(const float* __restrict__ X,
                                              unsigned short* __restrict__ Y, int n){
  int i = blockIdx.x*256 + threadIdx.x;
  if (i < n) Y[i] = f2bf(X[i]);
}

// ---------------- generic WMMA GEMM: C[M,N] = A[M,K] * W^T + bias ----------------
// block = 256 threads (8 waves), block tile 16(M) x 128(N); wave owns one 16-wide N tile
__global__ void __launch_bounds__(256) k_gemm(const unsigned short* __restrict__ A, int K,
                                              const unsigned short* __restrict__ Wf,
                                              const float* __restrict__ bias,
                                              float* __restrict__ Cf,
                                              unsigned short* __restrict__ Cbf,
                                              int N, int relu){
  int lane = threadIdx.x & 31, wave = threadIdx.x >> 5;
  int row0 = blockIdx.x << 4;
  int ntile = (blockIdx.y << 3) + wave;
  int ktiles = K >> 5;
  v8f acc = vzero();
  for (int kt=0; kt<ktiles; ++kt){
    v16bf a = afrag_g(A, K, row0, kt<<5, lane);
    v16bf b = bfrag_w(Wf, K, ntile, kt, lane);
    acc = WMMA_BF16(a, b, acc);
  }
  int nc = (ntile<<4) + (lane & 15);
  int half = lane >> 4;
  float bv = bias ? bias[nc] : 0.f;
#pragma unroll
  for (int r=0;r<8;r++){
    float v = acc[r] + bv;
    if (relu) v = fmaxf(v, 0.f);
    size_t m = (size_t)(row0 + r + (half<<3));
    if (Cf)  Cf [m*N + nc] = v;
    if (Cbf) Cbf[m*N + nc] = f2bf(v);
  }
}

// ---------------- row LayerNorm (H=256), optional residual add / tanh ----------------
__global__ void __launch_bounds__(256) k_ln(const float* __restrict__ X,
                                            const float* __restrict__ Add,
                                            const float* __restrict__ g,
                                            const float* __restrict__ be,
                                            float* __restrict__ outf,
                                            unsigned short* __restrict__ outb,
                                            int do_tanh){
  int row  = (blockIdx.x<<3) + (threadIdx.x>>5);
  int lane = threadIdx.x & 31;
  const float* x = X + (size_t)row*256;
  float v[8]; float s = 0.f;
#pragma unroll
  for (int i=0;i<8;i++){
    int c = lane + (i<<5);
    float t = x[c];
    if (Add) t += Add[(size_t)row*256 + c];
    v[i]=t; s+=t;
  }
#pragma unroll
  for (int o=16;o>0;o>>=1) s += __shfl_xor(s, o, 32);
  float mean = s * (1.f/256.f);
  float q = 0.f;
#pragma unroll
  for (int i=0;i<8;i++){ float d=v[i]-mean; q+=d*d; }
#pragma unroll
  for (int o=16;o>0;o>>=1) q += __shfl_xor(q, o, 32);
  float inv = rsqrtf(q*(1.f/256.f) + 1e-5f);
#pragma unroll
  for (int i=0;i<8;i++){
    int c = lane + (i<<5);
    float y = (v[i]-mean)*inv*g[c] + be[c];
    if (do_tanh) y = tanhf(y);
    if (outf) outf[(size_t)row*256 + c] = y;
    if (outb) outb[(size_t)row*256 + c] = f2bf(y);
  }
}

// ---------------- static small heads (fu / nv / sproj / sc / scat) ----------------
__global__ void __launch_bounds__(256) k_heads(const float* __restrict__ sh_f,
    const float* fu_w1,const float* fu_b1,const float* fu_w2,const float* fu_b2,
    const float* nv_w1,const float* nv_b1,const float* nv_w2,const float* nv_b2,
    const float* sc_w,const float* sc_b,const float* scat_w,const float* scat_b,
    const float* sproj_w,const float* sproj_b,
    float* out_fu, float* out_nv, float* out_scont, float* out_cat, float* out_cats,
    float* scond, float* nvr){
  int b = blockIdx.x, tid = threadIdx.x;
  __shared__ float sh[256];
  __shared__ float hid[128];
  __shared__ float red[128];
  __shared__ float scat[8];
  sh[tid] = sh_f[(size_t)b*256 + tid];
  __syncthreads();
  if (tid < 128){
    float a = fu_b1[tid];
    for (int k=0;k<256;k++) a += sh[k]*fu_w1[tid*256+k];
    hid[tid] = fmaxf(a, 0.f);
  }
  __syncthreads();
  if (tid < 128) red[tid] = hid[tid]*fu_w2[tid];
  __syncthreads();
  for (int s=64;s>0;s>>=1){ if (tid<s) red[tid]+=red[tid+s]; __syncthreads(); }
  if (tid==0) out_fu[b] = sigmoidf_(red[0] + fu_b2[0]);
  __syncthreads();
  if (tid < 128){
    float a = nv_b1[tid];
    for (int k=0;k<256;k++) a += sh[k]*nv_w1[tid*256+k];
    hid[tid] = fmaxf(a, 0.f);
  }
  __syncthreads();
  if (tid < 128) red[tid] = hid[tid]*nv_w2[tid];
  __syncthreads();
  for (int s=64;s>0;s>>=1){ if (tid<s) red[tid]+=red[tid+s]; __syncthreads(); }
  if (tid==0){
    float nv = softplusf_(red[0] + nv_b2[0]) + 1.f;
    nv = fminf(fmaxf(nv, 2.f), 256.f);
    out_nv[b] = nv;
    nvr[b] = fminf(fmaxf(rintf(nv), 2.f), 256.f);
  }
  if (tid < 16){
    float a = sproj_b[tid];
    for (int k=0;k<256;k++) a += sh[k]*sproj_w[tid*256+k];
    scond[(size_t)b*16 + tid] = a;
  } else if (tid < 20){
    int o = tid-16;
    float a = sc_b[o];
    for (int k=0;k<256;k++) a += sh[k]*sc_w[o*256+k];
    out_scont[b*4+o] = a;
  } else if (tid < 26){
    int o = tid-20;
    float a = scat_b[o];
    for (int k=0;k<256;k++) a += sh[k]*scat_w[o*256+k];
    scat[o] = a;
  }
  __syncthreads();
  if (tid==0){
    float ls[6]; int am=0; float best=-1e30f;
    for (int j=0;j<6;j++){
      float lh = scat[j] + gumbelr(1u,(unsigned)b,(unsigned)j);
      ls[j]    = scat[j] + gumbelr(2u,(unsigned)b,(unsigned)j);
      if (lh > best){ best = lh; am = j; }
    }
    for (int j=0;j<6;j++) out_cat[b*6+j] = (j==am) ? 1.f : 0.f;   // straight-through fwd value
    float mx=-1e30f; for (int j=0;j<6;j++) mx=fmaxf(mx,ls[j]);
    float s=0.f; for (int j=0;j<6;j++) s += __expf(ls[j]-mx);
    for (int j=0;j<6;j++) out_cats[b*6+j] = __expf(ls[j]-mx)/s;
  }
}

// ---------------- GRU cell: register-resident gates, WMMA bf16 ----------------
// wave w owns hidden cols [32w,32w+32): its 6 N-tiles are the r/z/n slices of those cols.
// A operands come from bf16 LDS copies (2x ds_load_b128 per fragment); the f32 copy of H
// is kept for the z*h_old term and for the f32 h_seq emission.
__device__ void gru_cell(const unsigned short* Ainb, int lda, int kta,
                         float* H, unsigned short* Hb,
                         const unsigned short* Wi, int KpadI,
                         const unsigned short* Wh,
                         const float* bih, const float* bhh,
                         int wave, int lane){
  v8f ai[6], ah[6];
#pragma unroll
  for (int j=0;j<6;j++){ ai[j]=vzero(); ah[j]=vzero(); }
  int ntb = wave << 1;
  int nt[6] = { ntb, ntb+1, 16+ntb, 17+ntb, 32+ntb, 33+ntb };
  for (int kt=0; kt<kta; ++kt){
    v16bf a = afrag_lb(Ainb, lda, kt<<5, lane);
#pragma unroll
    for (int j=0;j<6;j++) ai[j] = WMMA_BF16(a, bfrag_w(Wi, KpadI, nt[j], kt, lane), ai[j]);
  }
  for (int kt=0; kt<8; ++kt){
    v16bf a = afrag_lb(Hb, 256, kt<<5, lane);
#pragma unroll
    for (int j=0;j<6;j++) ah[j] = WMMA_BF16(a, bfrag_w(Wh, 256, nt[j], kt, lane), ah[j]);
  }
  int half = lane >> 4;
  float hnew[16];
#pragma unroll
  for (int j2=0;j2<2;j2++){
    int c = ((ntb+j2)<<4) + (lane & 15);
    float b_r  = bih[c]     + bhh[c];
    float b_z  = bih[256+c] + bhh[256+c];
    float b_in = bih[512+c];
    float b_hn = bhh[512+c];
#pragma unroll
    for (int r=0;r<8;r++){
      int m = r + (half<<3);
      float rg = sigmoidf_(ai[j2][r]   + ah[j2][r]   + b_r);
      float zg = sigmoidf_(ai[2+j2][r] + ah[2+j2][r] + b_z);
      float ng = tanhf(ai[4+j2][r] + b_in + rg*(ah[4+j2][r] + b_hn));
      float hold = H[m*256 + c];
      hnew[j2*8+r] = (1.f-zg)*ng + zg*hold;
    }
  }
  __syncthreads();          // all waves done reading H/Hb
#pragma unroll
  for (int j2=0;j2<2;j2++){
    int c = ((ntb+j2)<<4) + (lane & 15);
#pragma unroll
    for (int r=0;r<8;r++){
      int m = r + (half<<3);
      H [m*256 + c] = hnew[j2*8+r];
      Hb[m*256 + c] = f2bf(hnew[j2*8+r]);
    }
  }
  __syncthreads();
}

// ---------------- GRU scan: 16 persistent WGs, one 16-batch tile each ----------------
__global__ void __launch_bounds__(256) k_gru(
    const float* __restrict__ ztemp, const float* __restrict__ h0f,
    const float* __restrict__ scond,
    const unsigned short* __restrict__ Wih0, const unsigned short* __restrict__ Whh0,
    const unsigned short* __restrict__ Wih1, const unsigned short* __restrict__ Whh1,
    const float* __restrict__ bih0, const float* __restrict__ bhh0,
    const float* __restrict__ bih1, const float* __restrict__ bhh1,
    const unsigned short* __restrict__ Wtc1, const float* __restrict__ tc_b1,
    const float* __restrict__ tc_w2, const float* __restrict__ tc_b2,
    const unsigned short* __restrict__ Wint1, const float* __restrict__ int_b1,
    const float* __restrict__ int_w2, const float* __restrict__ int_b2,
    const float* __restrict__ cirr_w, const float* __restrict__ cirr_b,
    const float* __restrict__ creg_w, const float* __restrict__ creg_b,
    float* __restrict__ out_delta, float* __restrict__ out_cont,
    float* __restrict__ out_irr, float* __restrict__ out_reg,
    float* __restrict__ hseqf, unsigned short* __restrict__ hseqb){
  __shared__ float sh_h1[16][256];
  __shared__ float sh_h2[16][256];
  __shared__ unsigned short sh_h1b[16][256];   // bf16 copies: WMMA A operands
  __shared__ unsigned short sh_h2b[16][256];
  __shared__ unsigned short sh_ginb[16][96];   // gin only consumed as A operand
  __shared__ float sh_hid[16][128];
  __shared__ float sh_hid2[16][64];
  __shared__ float sh_xpc[16][8];
  __shared__ float sh_cpo[16][7];
  __shared__ float sh_dp[16];
  __shared__ float sh_reg[16][5];

  int tid = threadIdx.x, lane = tid & 31, wave = tid >> 5;
  int b0 = blockIdx.x << 4;

  for (int i = tid; i < 16*256; i += 256){
    int m = i >> 8, c = i & 255;
    float a = h0f[(size_t)(b0+m)*512 + c];
    float b = h0f[(size_t)(b0+m)*512 + 256 + c];
    sh_h1[m][c] = a; sh_h1b[m][c] = f2bf(a);
    sh_h2[m][c] = b; sh_h2b[m][c] = f2bf(b);
  }
  { int m = tid >> 4, j = tid & 15;
    sh_ginb[m][64+j] = f2bf(scond[(size_t)(b0+m)*16 + j]);
    sh_ginb[m][80+j] = 0; }
  if (tid < 128) sh_xpc[tid>>3][tid&7] = 0.f;
  if (tid < 112) sh_cpo[tid/7][tid%7] = 0.f;
  if (tid < 16)  sh_dp[tid] = 0.f;
  __syncthreads();

  for (int t = 0; t < T_; ++t){
    // build gin = [zt | xpc | cpo | dp | s_cond | pad] (bf16, A operand only)
    for (int i = tid; i < 16*64; i += 256){
      int m = i >> 6, c = i & 63;
      float v;
      if (c < 48)      v = ztemp[((size_t)(b0+m)*T_ + t)*48 + c];
      else if (c < 56) v = sh_xpc[m][c-48];
      else if (c < 63) v = sh_cpo[m][c-56];
      else             v = sh_dp[m];
      sh_ginb[m][c] = f2bf(v);
    }
    __syncthreads();

    gru_cell(&sh_ginb[0][0], 96, 3, &sh_h1[0][0], &sh_h1b[0][0], Wih0, 96,  Whh0, bih0, bhh0, wave, lane);
    gru_cell(&sh_h1b[0][0], 256, 8, &sh_h2[0][0], &sh_h2b[0][0], Wih1, 256, Whh1, bih1, bhh1, wave, lane);

    // emit h_t (f32 for residual/LN, bf16 as GEMM A input)
    for (int i = tid; i < 16*256; i += 256){
      int m = i >> 8, c = i & 255;
      size_t o = ((size_t)(b0+m)*T_ + t)*256 + c;
      hseqf[o] = sh_h2[m][c];
      hseqb[o] = sh_h2b[m][c];
    }

    // head hidden layers via WMMA (tc: N=128, int: N=64)
    {
      v8f at = vzero(), an = vzero();
      for (int kt=0; kt<8; ++kt){
        v16bf a = afrag_lb(&sh_h2b[0][0], 256, kt<<5, lane);
        at = WMMA_BF16(a, bfrag_w(Wtc1, 256, wave, kt, lane), at);
        if (wave < 4) an = WMMA_BF16(a, bfrag_w(Wint1, 256, wave, kt, lane), an);
      }
      int half = lane >> 4, nc = (wave<<4) + (lane & 15);
      float bt = tc_b1[nc];
#pragma unroll
      for (int r=0;r<8;r++) sh_hid[r+(half<<3)][nc] = fmaxf(at[r]+bt, 0.f);
      if (wave < 4){
        float bi = int_b1[nc];
#pragma unroll
        for (int r=0;r<8;r++) sh_hid2[r+(half<<3)][nc] = fmaxf(an[r]+bi, 0.f);
      }
    }
    __syncthreads();

    // small head dots
    if (tid < 128){
      int m = tid>>3, o = tid&7;
      float a = tc_b2[o];
      for (int k=0;k<128;k++) a += sh_hid[m][k]*tc_w2[o*128+k];
      out_cont[((size_t)(b0+m)*T_+t)*8 + o] = a;
      sh_xpc[m][o] = a;
    } else if (tid < 144){
      int m = tid-128;
      float a = int_b2[0];
      for (int k=0;k<64;k++) a += sh_hid2[m][k]*int_w2[k];
      float d = softplusf_(a);
      out_delta[(size_t)(b0+m)*T_ + t] = d;
      sh_dp[m] = d;
    } else if (tid < 160){
      int m = tid-144;
      float a = cirr_b[0];
      for (int k=0;k<256;k++) a += sh_h2[m][k]*cirr_w[k];
      float pi = sigmoidf_(a);
      size_t o = ((size_t)(b0+m)*T_+t)*2;
      out_irr[o] = 1.f-pi; out_irr[o+1] = pi;
      sh_cpo[m][0] = 1.f-pi; sh_cpo[m][1] = pi;
    } else if (tid < 240){
      int q = tid-160; int m = q/5, j = q-5*m;
      float a = creg_b[j];
      for (int k=0;k<256;k++) a += sh_h2[m][k]*creg_w[j*256+k];
      out_reg[((size_t)(b0+m)*T_+t)*5 + j] = a;
      sh_reg[m][j] = a;
    }
    __syncthreads();
    if (tid < 16){
      int m = tid;
      float l[5]; float mx = -1e30f;
      for (int j=0;j<5;j++){
        l[j] = sh_reg[m][j] + gumbelr(7777u, (unsigned)(t*B_ + b0 + m), (unsigned)j);
        mx = fmaxf(mx, l[j]);
      }
      float s = 0.f;
      for (int j=0;j<5;j++){ l[j] = __expf(l[j]-mx); s += l[j]; }
      for (int j=0;j<5;j++) sh_cpo[m][2+j] = l[j]/s;
    }
    __syncthreads();
    if (tid == 0 && t+1 < T_)
      __builtin_prefetch(&ztemp[((size_t)b0*T_ + (t+1))*48], 0, 1);  // global_prefetch_b8
  }
}

// ---------------- attention: one wave per (b, q-tile, head), flash online softmax ----------------
__global__ void __launch_bounds__(32) k_attn(const unsigned short* __restrict__ qkv,
                                             const float* __restrict__ nvr,
                                             unsigned short* __restrict__ ctxb){
  __shared__ unsigned short sh_pb[16][32];   // P tile staged directly in bf16
  int lane = threadIdx.x & 31;
  int b = blockIdx.y;
  int qt = blockIdx.x >> 2, h = blockIdx.x & 3;
  int nv = (int)nvr[b];
  int qrow = b*T_ + (qt<<4);
  int half = lane >> 4;

  v16bf aq0 = afrag_g(qkv, 768, qrow, h*64,      lane);
  v16bf aq1 = afrag_g(qkv, 768, qrow, h*64 + 32, lane);

  float rmax[8], rsum[8];
  v8f cacc[4];
#pragma unroll
  for (int r=0;r<8;r++){ rmax[r] = -1e30f; rsum[r] = 0.f; }
#pragma unroll
  for (int d=0;d<4;d++) cacc[d] = vzero();

  for (int jc=0; jc<8; ++jc){
    int j0 = jc << 5;
    if (j0 >= nv) break;                       // uniform per block
    v8f s0 = vzero(), s1 = vzero();
    s0 = WMMA_BF16(aq0, bfrag_rows(qkv,768, b*T_+j0,    256+h*64,    lane), s0);
    s0 = WMMA_BF16(aq1, bfrag_rows(qkv,768, b*T_+j0,    256+h*64+32, lane), s0);
    s1 = WMMA_BF16(aq0, bfrag_rows(qkv,768, b*T_+j0+16, 256+h*64,    lane), s1);
    s1 = WMMA_BF16(aq1, bfrag_rows(qkv,768, b*T_+j0+16, 256+h*64+32, lane), s1);

    int jA = j0 + (lane & 15), jB = j0 + 16 + (lane & 15);
#pragma unroll
    for (int r=0;r<8;r++){
      float a  = (jA < nv) ? s0[r]*0.125f : -1e9f;
      float c2 = (jB < nv) ? s1[r]*0.125f : -1e9f;
      float mrow = fmaxf(a, c2);
#pragma unroll
      for (int o=1;o<16;o<<=1) mrow = fmaxf(mrow, __shfl_xor(mrow, o, 32));
      float nm = fmaxf(rmax[r], mrow);
      float scale = __expf(rmax[r] - nm);
      float p0 = __expf(a - nm), p1 = __expf(c2 - nm);
      float ps = p0 + p1;
#pragma unroll
      for (int o=1;o<16;o<<=1) ps += __shfl_xor(ps, o, 32);
      rsum[r] = rsum[r]*scale + ps;
      rmax[r] = nm;
#pragma unroll
      for (int d=0;d<4;d++) cacc[d][r] *= scale;   // same (r,half) row mapping as stats
      sh_pb[r+(half<<3)][lane & 15]        = f2bf(p0);
      sh_pb[r+(half<<3)][16 + (lane & 15)] = f2bf(p1);
    }
    __syncthreads();
    v16bf ap = afrag_lb(&sh_pb[0][0], 32, 0, lane);
#pragma unroll
    for (int d=0;d<4;d++){
      v16bf bv = bfrag_cols(qkv, 768, b*T_+j0, 512 + h*64 + (d<<4), lane);
      cacc[d] = WMMA_BF16(ap, bv, cacc[d]);
    }
    __syncthreads();
  }
#pragma unroll
  for (int d=0;d<4;d++)
#pragma unroll
    for (int r=0;r<8;r++){
      int m = r + (half<<3);
      float v = cacc[d][r] / rsum[r];
      ctxb[((size_t)(qrow+m))*256 + h*64 + (d<<4) + (lane & 15)] = f2bf(v);
    }
}

// ---------------- final tc_head second layer (N=8, VALU dots) ----------------
__global__ void __launch_bounds__(256) k_tc2(const float* __restrict__ hid,
                                             const float* __restrict__ w2,
                                             const float* __restrict__ b2,
                                             float* __restrict__ out){
  int gid = blockIdx.x*256 + threadIdx.x;
  int row = gid >> 3, o = gid & 7;
  const float* hrow = hid + (size_t)row*128;
  const float* w = w2 + o*128;
  float a = b2[o];
  for (int k=0;k<128;k++) a += hrow[k]*w[k];
  out[(size_t)row*8 + o] = a;
}

// ---------------- host-side orchestration ----------------
extern "C" void kernel_launch(void* const* d_in, const int* in_sizes, int n_in,
                              void* d_out, int out_size, void* d_ws, size_t ws_size,
                              hipStream_t stream){
  (void)in_sizes; (void)n_in; (void)out_size; (void)ws_size;
  const float* z_static   = (const float*)d_in[0];
  const float* z_temporal = (const float*)d_in[1];
  const float* P[46];
  for (int i=0;i<46;i++) P[i] = (const float*)d_in[2+i];
  float* out = (float*)d_out;

  unsigned char* base = (unsigned char*)d_ws;
  size_t off = 0;
  auto abf = [&](size_t n)->unsigned short*{
    unsigned short* p = (unsigned short*)(base+off); off += ((n*2 + 255) & ~(size_t)255); return p; };
  auto af = [&](size_t n)->float*{
    float* p = (float*)(base+off); off += ((n*4 + 255) & ~(size_t)255); return p; };

  // bf16 fragment weights
  unsigned short* Wfc   = abf(256*64);
  unsigned short* Wth0  = abf(512*256);
  unsigned short* Wih0  = abf(768*96);
  unsigned short* Whh0  = abf(768*256);
  unsigned short* Wih1  = abf(768*256);
  unsigned short* Whh1  = abf(768*256);
  unsigned short* Watti = abf(768*256);
  unsigned short* Watto = abf(256*256);
  unsigned short* Wtc1  = abf(128*256);
  unsigned short* Wint1 = abf(64*256);
  // bf16 activations
  unsigned short* zsb   = abf(256*64);
  unsigned short* shb   = abf(256*256);
  unsigned short* hseqb = abf((size_t)65536*256);
  unsigned short* qkvb  = abf((size_t)65536*768);
  unsigned short* ctxb  = abf((size_t)65536*256);
  unsigned short* hlnb  = abf((size_t)65536*256);
  // f32 buffers
  float* pre   = af(256*256);
  float* shf   = af(256*256);
  float* h0f   = af(256*512);
  float* scnd  = af(256*16);
  float* nvr   = af(256);
  float* hseqf = af((size_t)65536*256);
  float* attof = af((size_t)65536*256);
  float* hidf  = af((size_t)65536*128);

  auto frag = [&](const float* W, unsigned short* Wf, int N, int Ko, int Kp){
    int tot = N*Kp;
    k_wfrag<<<(tot+255)/256, 256, 0, stream>>>(W, Wf, N, Ko, Kp);
  };
  frag(P[0],  Wfc,  256, 64,  64);
  frag(P[4],  Wth0, 512, 256, 256);
  frag(P[20], Wih0, 768, 80,  96);   // K padded 80 -> 96
  frag(P[21], Whh0, 768, 256, 256);
  frag(P[24], Wih1, 768, 256, 256);
  frag(P[25], Whh1, 768, 256, 256);
  frag(P[28], Watti,768, 256, 256);
  frag(P[30], Watto,256, 256, 256);
  frag(P[34], Wtc1, 128, 256, 256);
  frag(P[38], Wint1, 64, 256, 256);
  k_tobf<<<(256*64+255)/256, 256, 0, stream>>>(z_static, zsb, 256*64);

  // s_h = tanh(LN(z_static @ Wfc^T + b))
  k_gemm<<<dim3(16,2), 256, 0, stream>>>(zsb, 64, Wfc, P[1], pre, nullptr, 256, 0);
  k_ln<<<32, 256, 0, stream>>>(pre, nullptr, P[2], P[3], shf, shb, 1);
  // h0 = s_h @ Wto_h0^T + b
  k_gemm<<<dim3(16,4), 256, 0, stream>>>(shb, 256, Wth0, P[5], h0f, nullptr, 512, 0);
  // static heads
  k_heads<<<256, 256, 0, stream>>>(shf, P[6],P[7],P[8],P[9], P[10],P[11],P[12],P[13],
                                   P[14],P[15],P[16],P[17], P[18],P[19],
                                   out+O4, out+O5, out+O6, out+O7, out+O8, scnd, nvr);
  // GRU scan
  k_gru<<<16, 256, 0, stream>>>(z_temporal, h0f, scnd,
                                Wih0, Whh0, Wih1, Whh1,
                                P[22], P[23], P[26], P[27],
                                Wtc1, P[35], P[36], P[37],
                                Wint1, P[39], P[40], P[41],
                                P[42], P[43], P[44], P[45],
                                out+O2, out+O3, out+O9, out+O10,
                                hseqf, hseqb);
  // qkv = h_seq @ Wattn_in^T + b  (bf16 for attention fragments)
  k_gemm<<<dim3(4096,6), 256, 0, stream>>>(hseqb, 256, Watti, P[29], nullptr, qkvb, 768, 0);
  // attention -> ctx (bf16)
  k_attn<<<dim3(64,256), 32, 0, stream>>>(qkvb, nvr, ctxb);
  // attn_out = ctx @ Wattn_out^T + b
  k_gemm<<<dim3(4096,2), 256, 0, stream>>>(ctxb, 256, Watto, P[31], attof, nullptr, 256, 0);
  // h_seq = LN(h_seq + attn_out)
  k_ln<<<8192, 256, 0, stream>>>(hseqf, attof, P[32], P[33], nullptr, hlnb, 0);
  // temporal_cont = tc_head(h_seq): relu GEMM then 8-wide dots
  k_gemm<<<dim3(4096,1), 256, 0, stream>>>(hlnb, 256, Wtc1, P[35], hidf, nullptr, 128, 1);
  k_tc2<<<2048, 256, 0, stream>>>(hidf, P[36], P[37], out+O1);
}
